// RecurrentGCN_70781061038141
// MI455X (gfx1250) — matrix operand order repaired and checked by reading, not compiled, not run on hardware
//
#include <hip/hip_runtime.h>

// ---------------------------------------------------------------------------
// GConvLSTM (ChebConv K=1) fused kernel for MI455X / gfx1250.
// Per wave: 16 nodes batched as WMMA M-dim; bf16 WMMA 16x16x32 for all GEMMs.
// Round 3: v_perm-based bf16 packing, gate bias folded into the GEMM via a
//          1.0 column in the zero-padded K region (k=40).
// ---------------------------------------------------------------------------

typedef __attribute__((ext_vector_type(16))) __bf16 v16bf;
typedef __attribute__((ext_vector_type(8)))  float  v8f;

union V16 { v16bf v; unsigned u[8]; };

// --- fast math helpers -----------------------------------------------------

__device__ __forceinline__ float rcp_f(float x) {
#if __has_builtin(__builtin_amdgcn_rcpf)
  return __builtin_amdgcn_rcpf(x);          // v_rcp_f32
#else
  return 1.0f / x;
#endif
}

__device__ __forceinline__ float tanh_f(float x) {
#if __has_builtin(__builtin_amdgcn_tanhf)
  return __builtin_amdgcn_tanhf(x);         // gfx1250 v_tanh_f32
#elif __has_builtin(__builtin_amdgcn_tanh_f32)
  return __builtin_amdgcn_tanh_f32(x);
#else
  // tanh(x) = 1 - 2/(1 + e^{2x}); saturates correctly at +-inf
  return fmaf(-2.0f, rcp_f(1.0f + __expf(2.0f * x)), 1.0f);
#endif
}

__device__ __forceinline__ float sigmoid_f(float x) {
  return rcp_f(1.0f + __expf(-x));          // v_mul + v_exp + v_add + v_rcp
}

// pack two f32 into two bf16 (round-half-up) as one u32: 3 VALU (2 add + perm)
__device__ __forceinline__ unsigned pack_bf16(float lo, float hi) {
#if __has_builtin(__builtin_amdgcn_cvt_pk_bf16_f32)
  auto t = __builtin_amdgcn_cvt_pk_bf16_f32(lo, hi);   // v_cvt_pk_bf16_f32
  return __builtin_bit_cast(unsigned, t);
#else
  unsigned ul = __builtin_bit_cast(unsigned, lo) + 0x8000u;
  unsigned uh = __builtin_bit_cast(unsigned, hi) + 0x8000u;
  return __builtin_amdgcn_perm(uh, ul, 0x07060302u);   // {uh[31:16], ul[31:16]}
#endif
}

__device__ __forceinline__ unsigned short bf16_hi(float f) {
  unsigned u = __builtin_bit_cast(unsigned, f) + 0x8000u;
  return (unsigned short)(u >> 16);                    // selectable as d16_hi store
}

#define NODES_PER_BLOCK 128   // 8 waves x 16 nodes

__global__ __launch_bounds__(256)
void gconv_lstm_wmma(
    const float* __restrict__ x, const float* __restrict__ h, const float* __restrict__ c,
    const float* __restrict__ Wxi, const float* __restrict__ bxi, const float* __restrict__ Whi,
    const float* __restrict__ bhi, const float* __restrict__ bi,  const float* __restrict__ wci,
    const float* __restrict__ Wxf, const float* __restrict__ bxf, const float* __restrict__ Whf,
    const float* __restrict__ bhf, const float* __restrict__ bfv, const float* __restrict__ wcf,
    const float* __restrict__ Wxc, const float* __restrict__ bxc, const float* __restrict__ Whc,
    const float* __restrict__ bhc, const float* __restrict__ bcv,
    const float* __restrict__ Wxo, const float* __restrict__ bxo, const float* __restrict__ Who,
    const float* __restrict__ bho, const float* __restrict__ bov, const float* __restrict__ wco,
    const float* __restrict__ Wl,  const float* __restrict__ bl,
    float* __restrict__ out, long long N)
{
  // LDS: transposed bf16 weights so each lane's B-fragment is a contiguous 32B read.
  // Gate matrix k layout: k<8: Wx ; 8..39: Wh ; k==40: fused bias (A supplies 1.0) ; else 0.
  __shared__ __align__(32) __bf16 sWT[4 * 32 * 64];        // [gate][col 0..31][k 0..63]
  __shared__ __align__(32) __bf16 sWTl[16 * 32];           // [col 0..15][k 0..31] (col>=9 zero)
  __shared__ __align__(32) unsigned short sH0[8 * 16 * 32];// per-wave relu(h0) bf16 staging
  __shared__ float sBl[16];                                // bl padded to 16

  const int tid = threadIdx.x;

  // ---- stage weights/biases into LDS (once per block), paired u32 writes ----
  {
    const float* Wx[4] = {Wxi, Wxf, Wxc, Wxo};
    const float* Wh[4] = {Whi, Whf, Whc, Who};
    const float* Bx[4] = {bxi, bxf, bxc, bxo};
    const float* Bh[4] = {bhi, bhf, bhc, bho};
    const float* Bg[4] = {bi,  bfv, bcv, bov};

    unsigned* sWTu  = (unsigned*)sWT;
    unsigned* sWTlu = (unsigned*)sWTl;

    // k-pairs (2k, 2k+1) never straddle the x|h|bias|pad boundaries (8, 40, 42 even)
    for (int idx = tid; idx < 4 * 32 * 32; idx += 256) {
      int g = idx >> 10, rem = idx & 1023, col = rem >> 5, k = (rem & 31) * 2;
      float lo = 0.0f, hi = 0.0f;
      if (k < 8)       { lo = Wx[g][k * 32 + col];       hi = Wx[g][(k + 1) * 32 + col]; }
      else if (k < 40) { lo = Wh[g][(k - 8) * 32 + col]; hi = Wh[g][(k - 7) * 32 + col]; }
      else if (k == 40){ lo = Bx[g][col] + Bh[g][col] + Bg[g][col]; }
      sWTu[idx] = pack_bf16(lo, hi);
    }
    for (int idx = tid; idx < 16 * 16; idx += 256) {
      int col = idx >> 4, k = (idx & 15) * 2;
      float lo = (col < 9) ? Wl[k * 9 + col] : 0.0f;
      float hi = (col < 9) ? Wl[(k + 1) * 9 + col] : 0.0f;
      sWTlu[idx] = pack_bf16(lo, hi);
    }
    if (tid < 16) sBl[tid] = (tid < 9) ? bl[tid] : 0.0f;
  }

  // prefetch next block's streaming inputs while staging completes
  {
    long long nb = ((long long)blockIdx.x + 1) * NODES_PER_BLOCK;
    if (nb < N) {
      __builtin_prefetch(x + nb * 8  + (tid << 2), 0, 1);
      __builtin_prefetch(h + nb * 32 + (tid << 4), 0, 1);
      __builtin_prefetch(c + nb * 32 + (tid << 4), 0, 1);
    }
  }
  __syncthreads();

  const int lane = tid & 31;
  const int wave = tid >> 5;
  const int nn   = lane & 15;        // column within tile (WMMA N / B,C,D lane index)
  const int hf   = lane >> 4;        // half-wave selector
  const long long base = (long long)blockIdx.x * NODES_PER_BLOCK + wave * 16;

  // ---- build A fragments: [x(8) | h(32) | 1.0 @k=40 | pad] as 16x64 bf16 ----
  // A layout (16-bit, 16x32): lane m%16, VGPR p: K=16*hf+2p+sub (contiguous, matches B)
  const long long nodeA = base + nn;
  const long long nA = (nodeA < N) ? nodeA : (N - 1);

  V16 a0, a1;
#pragma unroll
  for (int p = 0; p < 8; ++p) {
    const int Kl = 16 * hf + 2 * p;   // 0..31, even
    // K-step 0: K = Kl  -> x[K] for K<8 else h[K-8]
    {
      float2 t2;
      if (Kl < 8) t2 = *(const float2*)(x + nA * 8  + Kl);
      else        t2 = *(const float2*)(h + nA * 32 + (Kl - 8));
      a0.u[p] = pack_bf16(t2.x, t2.y);
    }
    // K-step 1: K = 32+Kl -> h[24+Kl] while Kl<8, else zero pad
    {
      unsigned v = 0u;
      if (Kl < 8) { float2 t2 = *(const float2*)(h + nA * 32 + 24 + Kl); v = pack_bf16(t2.x, t2.y); }
      a1.u[p] = v;
    }
  }
  // K=40 (hf=0, p=4, sub=0): A[m][40] = bf16(1.0) -> GEMM adds the staged bias row
  if (!hf) a1.u[4] = 0x00003F80u;

  // ---- gate pre-activations: 4 gates x 2 N-tiles, C = 0 (bias comes via k=40) ----
  const v8f zc = {};  // single shared zero C octet
  v8f acc[4][2];

#pragma unroll
  for (int g = 0; g < 4; ++g) {
#pragma unroll
    for (int t = 0; t < 2; ++t) {
      const int col = t * 16 + nn;
      // B layout: lane col, element j: K = 16*hf + j (+32 per K-step) -> contiguous in sWT
      const v16bf b0 = *(const v16bf*)(sWT + (g * 32 + col) * 64 + 16 * hf);
      const v16bf b1 = *(const v16bf*)(sWT + (g * 32 + col) * 64 + 32 + 16 * hf);
      acc[g][t] = __builtin_amdgcn_wmma_f32_16x16x32_bf16(
          false, a0.v, false, b0, (short)0, zc, false, false);
      acc[g][t] = __builtin_amdgcn_wmma_f32_16x16x32_bf16(
          false, a1.v, false, b1, (short)0, acc[g][t], false, false);
    }
  }

  // ---- elementwise LSTM math on D-layout accumulators; emit h0/c0; stage relu(h0) ----
  float* outY = out;
  float* outH = out + 9ll  * N;
  float* outC = out + 41ll * N;

#pragma unroll
  for (int t = 0; t < 2; ++t) {
    const int j = t * 16 + nn;
    const float wciv = wci[j], wcfv = wcf[j], wcov = wco[j];
#pragma unroll
    for (int r = 0; r < 8; ++r) {
      const long long node = base + r + 8 * hf;        // C/D layout: M = r + 8*(lane/16)
      const long long nc = (node < N) ? node : (N - 1);
      const float cv = c[nc * 32 + j];
      const float ig = sigmoid_f(fmaf(wciv, cv, acc[0][t][r]));
      const float fg = sigmoid_f(fmaf(wcfv, cv, acc[1][t][r]));
      const float tg = tanh_f(acc[2][t][r]);
      const float c0 = fmaf(fg, cv, ig * tg);
      const float og = sigmoid_f(fmaf(wcov, c0, acc[3][t][r]));
      const float h0 = og * tanh_f(c0);
      if (node < N) {
        outH[node * 32 + j] = h0;
        outC[node * 32 + j] = c0;
      }
      sH0[wave * 512 + (r + 8 * hf) * 32 + j] = bf16_hi(fmaxf(h0, 0.0f));
    }
  }

  // wave-private LDS tile: make the staged stores land before fragment reads
  asm volatile("s_wait_dscnt 0" ::: "memory");

  // ---- y = relu(h0) @ Wl + bl : one WMMA (K=32, OUT padded 9->16) ----
  V16 ah;
#pragma unroll
  for (int p = 0; p < 8; ++p) {
    const int K = 16 * hf + 2 * p;
    ah.u[p] = *(const unsigned*)(sH0 + wave * 512 + nn * 32 + K);
  }
  const v16bf blf = *(const v16bf*)(sWTl + nn * 32 + 16 * hf);
  v8f yacc;
  {
    const float bv = sBl[nn];
#pragma unroll
    for (int r = 0; r < 8; ++r) yacc[r] = bv;
  }
  yacc = __builtin_amdgcn_wmma_f32_16x16x32_bf16(
      false, ah.v, false, blf, (short)0, yacc, false, false);

  if (nn < 9) {
#pragma unroll
    for (int r = 0; r < 8; ++r) {
      const long long node = base + r + 8 * hf;
      if (node < N) outY[node * 9 + nn] = yacc[r];
    }
  }
}

extern "C" void kernel_launch(void* const* d_in, const int* in_sizes, int n_in,
                              void* d_out, int out_size, void* d_ws, size_t ws_size,
                              hipStream_t stream) {
  (void)n_in; (void)out_size; (void)d_ws; (void)ws_size;
  const float* x   = (const float*)d_in[0];
  // d_in[1] = edge_index (int64), d_in[2] = edge_weight: unused for ChebConv K=1
  const float* h   = (const float*)d_in[3];
  const float* c   = (const float*)d_in[4];
  const float* Wxi = (const float*)d_in[5];
  const float* bxi = (const float*)d_in[6];
  const float* Whi = (const float*)d_in[7];
  const float* bhi = (const float*)d_in[8];
  const float* bi  = (const float*)d_in[9];
  const float* wci = (const float*)d_in[10];
  const float* Wxf = (const float*)d_in[11];
  const float* bxf = (const float*)d_in[12];
  const float* Whf = (const float*)d_in[13];
  const float* bhf = (const float*)d_in[14];
  const float* bfv = (const float*)d_in[15];
  const float* wcf = (const float*)d_in[16];
  const float* Wxc = (const float*)d_in[17];
  const float* bxc = (const float*)d_in[18];
  const float* Whc = (const float*)d_in[19];
  const float* bhc = (const float*)d_in[20];
  const float* bcv = (const float*)d_in[21];
  const float* Wxo = (const float*)d_in[22];
  const float* bxo = (const float*)d_in[23];
  const float* Who = (const float*)d_in[24];
  const float* bho = (const float*)d_in[25];
  const float* bov = (const float*)d_in[26];
  const float* wco = (const float*)d_in[27];
  const float* Wl  = (const float*)d_in[28];
  const float* bl  = (const float*)d_in[29];
  float* out = (float*)d_out;

  const long long N = (long long)in_sizes[0] / 8;
  const int grid = (int)((N + NODES_PER_BLOCK - 1) / NODES_PER_BLOCK);

  gconv_lstm_wmma<<<dim3(grid), dim3(256), 0, stream>>>(
      x, h, c,
      Wxi, bxi, Whi, bhi, bi,  wci,
      Wxf, bxf, Whf, bhf, bfv, wcf,
      Wxc, bxc, Whc, bhc, bcv,
      Wxo, bxo, Who, bho, bov, wco,
      Wl, bl, out, N);
}
